// MoE_49074296324468
// MI455X (gfx1250) — matrix operand order repaired and checked by reading, not compiled, not run on hardware
//
#include <hip/hip_runtime.h>
#include <hip/hip_bf16.h>

// ---------------------------------------------------------------------------
// Problem constants (reference: B=8, S=8192, D=256, F=1024, E=8)
// ---------------------------------------------------------------------------
#define NTOK   65536          // B*S tokens
#define DDIM   256
#define FDIM   1024
#define NEXP   8
#define TILE_M 128
#define NTILES (NTOK / TILE_M)   // 512

typedef __attribute__((ext_vector_type(16))) __bf16 v16bf;
typedef __attribute__((ext_vector_type(8)))  float  v8f;
typedef __attribute__((ext_vector_type(4)))  unsigned int u32x4;
typedef __attribute__((ext_vector_type(4)))  int i32x4;
typedef __attribute__((ext_vector_type(8)))  int i32x8;

// LDS strides (elements). stride*2 bytes multiple of 16 so b128 frag loads
// stay aligned; padding reduces bank conflicts.
#define W1_STRIDE 264   // sW1[f 0..31][d 0..255]  row = 128 dw + 4 dw pad
#define W2_STRIDE 40    // sW2[d 0..255][f 0..31]  row = 16 dw  + 4 dw pad
#define H_STRIDE  40    // sH [r 0..127][f 0..31]

#if __has_builtin(__builtin_amdgcn_tensor_load_to_lds)
#define HAVE_TDM 1
#else
#define HAVE_TDM 0
#endif

// ---------------------------------------------------------------------------
// threefry2x32 (20 rounds) — Gumbel-max routing, approximating
// jax.random.categorical(jax.random.key(42), logits)
// ---------------------------------------------------------------------------
__device__ __forceinline__ unsigned tf_rotl(unsigned x, int n) {
    return (x << n) | (x >> (32 - n));
}

__device__ __forceinline__ void threefry2x32(unsigned k0, unsigned k1,
                                             unsigned c0, unsigned c1,
                                             unsigned* o0, unsigned* o1) {
    const unsigned ks2 = 0x1BD11BDAu ^ k0 ^ k1;
    unsigned x0 = c0 + k0, x1 = c1 + k1;
    const int rA0 = 13, rA1 = 15, rA2 = 26, rA3 = 6;
    const int rB0 = 17, rB1 = 29, rB2 = 16, rB3 = 24;
    x0 += x1; x1 = tf_rotl(x1, rA0); x1 ^= x0;
    x0 += x1; x1 = tf_rotl(x1, rA1); x1 ^= x0;
    x0 += x1; x1 = tf_rotl(x1, rA2); x1 ^= x0;
    x0 += x1; x1 = tf_rotl(x1, rA3); x1 ^= x0;
    x0 += k1; x1 += ks2 + 1u;
    x0 += x1; x1 = tf_rotl(x1, rB0); x1 ^= x0;
    x0 += x1; x1 = tf_rotl(x1, rB1); x1 ^= x0;
    x0 += x1; x1 = tf_rotl(x1, rB2); x1 ^= x0;
    x0 += x1; x1 = tf_rotl(x1, rB3); x1 ^= x0;
    x0 += ks2; x1 += k0 + 2u;
    x0 += x1; x1 = tf_rotl(x1, rA0); x1 ^= x0;
    x0 += x1; x1 = tf_rotl(x1, rA1); x1 ^= x0;
    x0 += x1; x1 = tf_rotl(x1, rA2); x1 ^= x0;
    x0 += x1; x1 = tf_rotl(x1, rA3); x1 ^= x0;
    x0 += k0; x1 += k1 + 3u;
    x0 += x1; x1 = tf_rotl(x1, rB0); x1 ^= x0;
    x0 += x1; x1 = tf_rotl(x1, rB1); x1 ^= x0;
    x0 += x1; x1 = tf_rotl(x1, rB2); x1 ^= x0;
    x0 += x1; x1 = tf_rotl(x1, rB3); x1 ^= x0;
    x0 += k1; x1 += ks2 + 4u;
    x0 += x1; x1 = tf_rotl(x1, rA0); x1 ^= x0;
    x0 += x1; x1 = tf_rotl(x1, rA1); x1 ^= x0;
    x0 += x1; x1 = tf_rotl(x1, rA2); x1 ^= x0;
    x0 += x1; x1 = tf_rotl(x1, rA3); x1 ^= x0;
    x0 += ks2; x1 += k0 + 5u;
    *o0 = x0; *o1 = x1;
}

__device__ __forceinline__ float gumbel_from_idx(unsigned idx) {
    const unsigned half = (NTOK * NEXP) / 2u;
    unsigned o0, o1, bits;
    if (idx < half) { threefry2x32(0u, 42u, idx, idx + half, &o0, &o1); bits = o0; }
    else            { threefry2x32(0u, 42u, idx - half, idx, &o0, &o1); bits = o1; }
    unsigned fb = (bits >> 9) | 0x3F800000u;
    float u = __uint_as_float(fb) - 1.0f;
    u = fmaxf(u, 1.17549435e-38f);
    return -logf(-logf(u));
}

// ---------------------------------------------------------------------------
// Kernel 0: zero routing counters
// ---------------------------------------------------------------------------
__global__ void moe_zero_counts(int* counts) {
    if (threadIdx.x < NEXP) counts[threadIdx.x] = 0;
}

// ---------------------------------------------------------------------------
// Kernel 1: weight transpose + fp32 -> bf16.  src: per-expert [M][N] fp32,
// dst: per-expert [N][M] bf16.  32x32 tiles via LDS, blockIdx.z = expert.
// ---------------------------------------------------------------------------
__global__ __launch_bounds__(256) void moe_prep_weights(
    const float* __restrict__ src, __bf16* __restrict__ dst, int M, int N) {
    __shared__ float t[32][33];
    const int e  = blockIdx.z;
    const int c0 = blockIdx.x * 32;
    const int r0 = blockIdx.y * 32;
    const int tx = threadIdx.x & 31;
    const int ty = threadIdx.x >> 5;           // 0..7
    const float* s = src + (size_t)e * M * N;
    __bf16*      d = dst + (size_t)e * M * N;
#pragma unroll
    for (int k = 0; k < 4; ++k)
        t[ty + 8 * k][tx] = s[(size_t)(r0 + ty + 8 * k) * N + c0 + tx];
    __syncthreads();
#pragma unroll
    for (int k = 0; k < 4; ++k)
        d[(size_t)(c0 + ty + 8 * k) * M + r0 + tx] = (__bf16)t[tx][ty + 8 * k];
}

// ---------------------------------------------------------------------------
// Kernel 2: gate logits + Gumbel-max routing.  One wave32 per token.
// ---------------------------------------------------------------------------
__global__ __launch_bounds__(256) void moe_gate_route(
    const float* __restrict__ x, const float* __restrict__ gw,
    const float* __restrict__ gb, int* __restrict__ counts,
    int* __restrict__ lists) {
    const int lane = threadIdx.x & 31;
    const int wid  = threadIdx.x >> 5;
    const int t    = blockIdx.x * 8 + wid;

    float p[NEXP];
#pragma unroll
    for (int e = 0; e < NEXP; ++e) p[e] = 0.0f;

    const float* xr = x + (size_t)t * DDIM;
    for (int d = lane; d < DDIM; d += 32) {
        const float xv = xr[d];
        const float* g = gw + d * NEXP;
#pragma unroll
        for (int e = 0; e < NEXP; ++e) p[e] += xv * g[e];
    }
#pragma unroll
    for (int off = 16; off > 0; off >>= 1) {
#pragma unroll
        for (int e = 0; e < NEXP; ++e) p[e] += __shfl_xor(p[e], off, 32);
    }

    int best = 0;
    float bv = -3.4e38f;
#pragma unroll
    for (int e = 0; e < NEXP; ++e) {
        const float v = p[e] + gb[e] + gumbel_from_idx((unsigned)(t * NEXP + e));
        if (v > bv) { bv = v; best = e; }
    }
    if (lane == 0) {
        const int pos = atomicAdd(counts + best, 1);
        lists[(size_t)best * NTOK + pos] = t;
    }
}

// ---------------------------------------------------------------------------
// TDM: async 2D tile (global bf16 -> LDS) with per-row LDS padding.
// Descriptor packing per CDNA5 ISA §8.3/8.4.  data_size = 4B (dword units).
// ---------------------------------------------------------------------------
#if HAVE_TDM
__device__ __forceinline__ void tdm_load_2d(
    unsigned lds_addr, const void* gptr,
    unsigned tile_d0_dw, unsigned tile_d1,
    unsigned tens_d0_dw, unsigned tens_d1, unsigned tens_d0_stride_dw,
    unsigned pad_interval_code, unsigned pad_amount_code) {
    unsigned long long ga = (unsigned long long)(uintptr_t)gptr;
    u32x4 g0;
    g0[0] = 1u;                                       // count=1, user mode
    g0[1] = lds_addr;                                 // LDS byte address
    g0[2] = (unsigned)(ga & 0xffffffffu);             // global addr [31:0]
    g0[3] = (unsigned)((ga >> 32) & 0x1ffffffu)       // global addr [56:32]
          | (2u << 30);                               // type = 2 ("image")
    i32x8 g1;
    g1[0] = (int)((2u << 16)                          // data_size: 4B
                | (1u << 20)                          // pad_enable
                | (pad_interval_code << 22)
                | (pad_amount_code << 25));
    g1[1] = (int)((tens_d0_dw & 0xffffu) << 16);      // tensor_dim0 lo16
    g1[2] = (int)(((tens_d0_dw >> 16) & 0xffffu)      // tensor_dim0 hi16
                | ((tens_d1 & 0xffffu) << 16));       // tensor_dim1 lo16
    g1[3] = (int)(((tens_d1 >> 16) & 0xffffu)         // tensor_dim1 hi16
                | ((tile_d0_dw & 0xffffu) << 16));    // tile_dim0
    g1[4] = (int)(tile_d1 & 0xffffu);                 // tile_dim1, tile_dim2=0
    g1[5] = (int)tens_d0_stride_dw;                   // dim0 stride [31:0]
    g1[6] = 0;                                        // stride hi, dim1 stride
    g1[7] = 0;
    i32x4 z4 = (i32x4){0, 0, 0, 0};
#if defined(__clang_major__) && (__clang_major__ >= 23)
    i32x8 z8 = (i32x8){0, 0, 0, 0, 0, 0, 0, 0};
    __builtin_amdgcn_tensor_load_to_lds(g0, g1, z4, z4, z8, 0);
#else
    __builtin_amdgcn_tensor_load_to_lds(g0, g1, z4, z4, 0);
#endif
}

template <int N>
__device__ __forceinline__ void tdm_wait() {
#if __has_builtin(__builtin_amdgcn_s_wait_tensorcnt)
    __builtin_amdgcn_s_wait_tensorcnt(N);   // immediate required
#else
    asm volatile("s_wait_tensorcnt %0" :: "i"(N) : "memory");
#endif
}
#endif  // HAVE_TDM

__device__ __forceinline__ unsigned lds_byte_addr(const void* p) {
    return (unsigned)(uintptr_t)p;   // flat LDS address: low 32 bits = offset
}

// ---------------------------------------------------------------------------
// LDS bf16 fragment load: 16 contiguous bf16 as 2x b128
// ---------------------------------------------------------------------------
__device__ __forceinline__ v16bf lds_frag16(const __bf16* p) {
    union { uint4 q[2]; v16bf v; } u;
    u.q[0] = *(const uint4*)(p);
    u.q[1] = *(const uint4*)(p + 8);
    return u.v;
}

// ---------------------------------------------------------------------------
// Kernel 3: fused grouped FFN.  grid = (NTILES, NEXP), 256 threads (8 waves).
// 128 gathered tokens -> relu(x@W1[e]+b1)@W2[e]+b2, bf16 WMMA, f32 accum.
// Weight slabs arrive in LDS via double-buffered TDM (TENSORcnt overlap).
// ---------------------------------------------------------------------------
__global__ __launch_bounds__(256) void moe_ffn(
    const float*  __restrict__ x,   const __bf16* __restrict__ w1p,  // [E][F][D]
    const float*  __restrict__ b1,  const __bf16* __restrict__ w2p,  // [E][D][F]
    const float*  __restrict__ b2,  const int*    __restrict__ counts,
    const int*    __restrict__ lists, float* __restrict__ out) {

    const int e    = blockIdx.y;
    const int tile = blockIdx.x;
    const int cnt  = counts[e];
    const int row0 = tile * TILE_M;
    if (row0 >= cnt) return;
    const int rows = min(TILE_M, cnt - row0);

    __shared__ alignas(16) int    sTok[TILE_M];
    __shared__ alignas(16) __bf16 sW1[2][32 * W1_STRIDE];    // [f][d]
    __shared__ alignas(16) __bf16 sW2[2][DDIM * W2_STRIDE];  // [d][f]
    __shared__ alignas(16) __bf16 sH [TILE_M * H_STRIDE];    // [row][f]

    const int tid    = threadIdx.x;
    const int lane   = tid & 31;
    const int wid    = tid >> 5;
    const int lan16  = lane & 15;
    const int hiHalf = (lane >= 16);
    const int kb     = hiHalf ? 8 : 0;   // 2*kb = A-fragment K base (contiguous k = 16h + s)

    if (tid < TILE_M)
        sTok[tid] = lists[(size_t)e * NTOK + row0 + (tid < rows ? tid : 0)];
    __syncthreads();

    // --- Resident x A-fragments: this wave's 16 rows, K = 0..255 (8 frags) ---
    const int myrow = wid * 16 + lan16;
    const float* xrow = x + (size_t)sTok[myrow] * DDIM;
    v16bf xf[8];
#pragma unroll
    for (int s = 0; s < 8; ++s) {
        const float* p0 = xrow + 32 * s + 2 * kb;
        const float* p1 = p0 + 8;
        float4 a0 = *(const float4*)(p0);
        float4 a1 = *(const float4*)(p0 + 4);
        float4 c0 = *(const float4*)(p1);
        float4 c1 = *(const float4*)(p1 + 4);
        v16bf f;
        f[0]=(__bf16)a0.x;  f[1]=(__bf16)a0.y;  f[2]=(__bf16)a0.z;  f[3]=(__bf16)a0.w;
        f[4]=(__bf16)a1.x;  f[5]=(__bf16)a1.y;  f[6]=(__bf16)a1.z;  f[7]=(__bf16)a1.w;
        f[8]=(__bf16)c0.x;  f[9]=(__bf16)c0.y;  f[10]=(__bf16)c0.z; f[11]=(__bf16)c0.w;
        f[12]=(__bf16)c1.x; f[13]=(__bf16)c1.y; f[14]=(__bf16)c1.z; f[15]=(__bf16)c1.w;
        xf[s] = f;
    }

    v8f acc[16];
#pragma unroll
    for (int nt = 0; nt < 16; ++nt) acc[nt] = (v8f){0,0,0,0,0,0,0,0};

    const __bf16* w1e = w1p + (size_t)e * FDIM * DDIM;   // [F][D]
    const __bf16* w2e = w2p + (size_t)e * DDIM * FDIM;   // [D][F]
    const float*  b1e = b1 + e * FDIM;
    const float*  b2e = b2 + e * DDIM;

#if HAVE_TDM
    // Stage slab ft=0 into buffer 0:
    //  sW1: 32 rows x 128 dw, pad 4 dw/row   (codes: interval 6, amount 3)
    //  sW2: 256 rows x 16 dw, pad 4 dw/row   (codes: interval 3, amount 3)
    if (wid == 0) {
        tdm_load_2d(lds_byte_addr(&sW1[0][0]), w1e, 128, 32,
                    128, 32, 128, 6u, 3u);
        tdm_load_2d(lds_byte_addr(&sW2[0][0]), w2e, 16, DDIM,
                    FDIM / 2, DDIM, FDIM / 2, 3u, 3u);
    }
#endif

    for (int ft = 0; ft < FDIM / 32; ++ft) {
        const int f0 = ft * 32;
        const int bb = ft & 1;

#if HAVE_TDM
        if (wid == 0) {
            if (ft + 1 < FDIM / 32) {     // issue next slab into other buffer
                const int f1 = f0 + 32;
                tdm_load_2d(lds_byte_addr(&sW1[bb ^ 1][0]),
                            w1e + (size_t)f1 * DDIM, 128, 32,
                            128, 32, 128, 6u, 3u);
                tdm_load_2d(lds_byte_addr(&sW2[bb ^ 1][0]),
                            w2e + f1, 16, DDIM,
                            FDIM / 2, DDIM, FDIM / 2, 3u, 3u);
                tdm_wait<2>();            // current slab done; next in flight
            } else {
                tdm_wait<0>();
            }
        }
#else
        // Fallback: cooperative b128 copy of pre-converted bf16 slabs.
#pragma unroll
        for (int it = 0; it < 4; ++it) {  // sW1: 32 rows x 512B
            const int lin = it * 256 + tid;
            const int r = lin >> 5, c = (lin & 31) * 8;
            *(uint4*)&sW1[bb][r * W1_STRIDE + c] =
                *(const uint4*)(w1e + (size_t)(f0 + r) * DDIM + c);
        }
#pragma unroll
        for (int it = 0; it < 4; ++it) {  // sW2: 256 rows x 64B
            const int lin = it * 256 + tid;
            const int r = lin >> 2, c = (lin & 3) * 8;
            *(uint4*)&sW2[bb][r * W2_STRIDE + c] =
                *(const uint4*)(w2e + (size_t)r * FDIM + f0 + c);
        }
#endif
        __syncthreads();   // slab bb visible to all waves

        // --- GEMM1: h[16 rows][32 f] = x(16x256) @ W1(256x32) ---
        v8f h0 = (v8f){0,0,0,0,0,0,0,0};
        v8f h1 = (v8f){0,0,0,0,0,0,0,0};
#pragma unroll
        for (int s = 0; s < 8; ++s) {
            const int koff = 32 * s + (hiHalf ? 16 : 0);
            v16bf bA = lds_frag16(&sW1[bb][(lan16)      * W1_STRIDE + koff]);
            v16bf bB = lds_frag16(&sW1[bb][(16 + lan16) * W1_STRIDE + koff]);
            h0 = __builtin_amdgcn_wmma_f32_16x16x32_bf16(false, xf[s], false, bA,
                                                         (short)0, h0, false, false);
            h1 = __builtin_amdgcn_wmma_f32_16x16x32_bf16(false, xf[s], false, bB,
                                                         (short)0, h1, false, false);
        }
        const float bias0 = b1e[f0 + lan16];
        const float bias1 = b1e[f0 + 16 + lan16];
#pragma unroll
        for (int i = 0; i < 8; ++i) {
            const int r = wid * 16 + i + (hiHalf ? 8 : 0);
            float v0 = h0[i] + bias0; v0 = v0 > 0.0f ? v0 : 0.0f;
            float v1 = h1[i] + bias1; v1 = v1 > 0.0f ? v1 : 0.0f;
            sH[r * H_STRIDE + lan16]      = (__bf16)v0;
            sH[r * H_STRIDE + 16 + lan16] = (__bf16)v1;
        }
        __syncthreads();   // sH ready

        // --- GEMM2 step: acc += h(16x32) @ W2slab(32x256) ---
        v16bf hf = lds_frag16(&sH[(wid * 16 + lan16) * H_STRIDE + (hiHalf ? 16 : 0)]);
        const int koffb = hiHalf ? 16 : 0;
#pragma unroll
        for (int nt = 0; nt < 16; ++nt) {
            v16bf bf = lds_frag16(&sW2[bb][(nt * 16 + lan16) * W2_STRIDE + koffb]);
            acc[nt] = __builtin_amdgcn_wmma_f32_16x16x32_bf16(false, hf, false, bf,
                                                              (short)0, acc[nt],
                                                              false, false);
        }
        __syncthreads();   // all reads of buffer bb done (TDM may overwrite)
    }

    // --- epilogue: + b2, scatter per-token rows ---
#pragma unroll
    for (int nt = 0; nt < 16; ++nt) {
        const float bias = b2e[nt * 16 + lan16];
#pragma unroll
        for (int i = 0; i < 8; ++i) {
            const int r = wid * 16 + i + (hiHalf ? 8 : 0);
            if (r < rows) {
                out[(size_t)sTok[r] * DDIM + nt * 16 + lan16] = acc[nt][i] + bias;
            }
        }
    }
}

// ---------------------------------------------------------------------------
// Launch.  d_ws layout:
//   [0, 256)              int counts[8]
//   [4096, 4096+2MB)      int lists[8][65536]
//   [2101248, +4MB)       bf16 w1p[E][F][D]
//   [6295552, +4MB)       bf16 w2p[E][D][F]
// ---------------------------------------------------------------------------
extern "C" void kernel_launch(void* const* d_in, const int* in_sizes, int n_in,
                              void* d_out, int out_size, void* d_ws, size_t ws_size,
                              hipStream_t stream) {
    (void)in_sizes; (void)n_in; (void)out_size; (void)ws_size;
    const float* x  = (const float*)d_in[0];
    const float* gw = (const float*)d_in[1];
    const float* gb = (const float*)d_in[2];
    const float* w1 = (const float*)d_in[3];
    const float* b1 = (const float*)d_in[4];
    const float* w2 = (const float*)d_in[5];
    const float* b2 = (const float*)d_in[6];
    float* out = (float*)d_out;

    int*    counts = (int*)d_ws;
    int*    lists  = (int*)((char*)d_ws + 4096);
    __bf16* w1p    = (__bf16*)((char*)d_ws + 2101248);
    __bf16* w2p    = (__bf16*)((char*)d_ws + 6295552);

    moe_zero_counts<<<1, 32, 0, stream>>>(counts);

    // Weight prep: transpose + bf16.  w1 [E][256][1024] -> w1p [E][1024][256]
    dim3 g1(FDIM / 32, DDIM / 32, NEXP);
    moe_prep_weights<<<g1, 256, 0, stream>>>(w1, w1p, DDIM, FDIM);
    // w2 [E][1024][256] -> w2p [E][256][1024]
    dim3 g2(DDIM / 32, FDIM / 32, NEXP);
    moe_prep_weights<<<g2, 256, 0, stream>>>(w2, w2p, FDIM, DDIM);

    moe_gate_route<<<NTOK / 8, 256, 0, stream>>>(x, gw, gb, counts, lists);

    dim3 grid(NTILES, NEXP);
    moe_ffn<<<grid, 256, 0, stream>>>(x, w1p, b1, w2p, b2, counts, lists, out);
}
